// MWT_CZ2d_18863496364218
// MI455X (gfx1250) — compile-verified, matrix operand order checked by reading, not run on hardware
//
// MWT_CZ2d forward for MI455X (gfx1250).
// Every stage = wave32 WMMA (v_wmma_f32_16x16x32_bf16) GEMM with fused
// gather/scatter. rfft2/irfft2 replaced by partial-DFT GEMMs (24x12 modes),
// twiddle tables generated on device (L2-resident). All operands staged so
// each lane's A/B fragment is two contiguous float4 pairs (global_load_b128),
// K always a multiple of 32 -> branchless WMMA inner loop (EXEC stays all-1s).
// 3-D grid (mtile, ntile, batch), shift-only index math, tile-level store
// epilogues (2 address decompositions per 16x16 tile).
#include <hip/hip_runtime.h>
#include <stdint.h>
#include <stddef.h>

typedef __attribute__((ext_vector_type(16))) __bf16 v16bf;
typedef __attribute__((ext_vector_type(8)))  float  v8f;

__device__ inline void ld8(const float* p, float o[8]) {
  float4 u = *(const float4*)p;
  float4 v = *(const float4*)(p + 4);
  o[0]=u.x; o[1]=u.y; o[2]=u.z; o[3]=u.w;
  o[4]=v.x; o[5]=v.y; o[6]=v.z; o[7]=v.w;
}

// ---------------------------------------------------------------------------
// One wave -> one 16x16 f32 tile. CDNA5 16-bit fragment swizzle:
//   lane<16 : row/col = lane,    elem e<8 -> k=k0+e,  e>=8 -> k=k0+16+(e-8)
//   lane>=16: row/col = lane-16, add 8 to k
// C (f32): vgpr r -> row = r + (lane>=16 ? 8 : 0), col = lane & 15.
// Grid: x = m-tiles (8 waves/block), y = n-tile, z = batch.
// ---------------------------------------------------------------------------
template<class AOp, class BOp, class COp>
__global__ __launch_bounds__(256) void wmma_gemm(int mtiles, int K,
                                                 AOp A, BOp B, COp C) {
  const int mt = blockIdx.x * (blockDim.x >> 5) + (threadIdx.x >> 5);
  if (mt >= mtiles) return;                  // wave-uniform tail
  const int nt = blockIdx.y, bat = blockIdx.z;
  const int lane = threadIdx.x & 31;
  const int rc  = lane & 15;
  const int hi8 = (lane & 16) >> 1;          // 0 or 8
  const int row = (mt << 4) + rc, col = (nt << 4) + rc;
  v8f acc = {0.f,0.f,0.f,0.f,0.f,0.f,0.f,0.f};
  for (int k0 = 0; k0 < K; k0 += 32) {       // K % 32 == 0 always
    float a0[8], a1[8], b0[8], b1[8];
    A.load8(bat, row, k0 + hi8,      a0);
    A.load8(bat, row, k0 + 16 + hi8, a1);
    B.load8(bat, k0 + hi8,      col, b0);
    B.load8(bat, k0 + 16 + hi8, col, b1);
    v16bf a, b;
#pragma unroll
    for (int e = 0; e < 8; ++e) {
      a[e] = (__bf16)a0[e]; a[e+8] = (__bf16)a1[e];
      b[e] = (__bf16)b0[e]; b[e+8] = (__bf16)b1[e];
    }
    acc = __builtin_amdgcn_wmma_f32_16x16x32_bf16(false, a, false, b,
                                                  (short)0, acc, false, false);
  }
  C.store_tile(bat, mt << 4, hi8, col, acc);
}

// ------------------------------ functors -----------------------------------
// Wavelet analysis: rows=(b,x,y,c), K=64 (4 quadrants x 16 ich)
struct WavA { const float* x; int logNin, logNh;
  __device__ void load8(int, int row, int k0, float o[8]) const {
    int mh = (1 << logNh) - 1;
    int c = row & 3; int t = row >> 2;
    int yo = t & mh; t >>= logNh; int xo = t & mh; int b = t >> logNh;
    int q = k0 >> 4, i0 = k0 & 15;
    int xx = 2*xo + (q >> 1), yy = 2*yo + (q & 1);
    size_t pix = ((((size_t)b << logNin) + xx) << logNin) + yy;
    ld8(x + (pix << 6) + (c << 4) + i0, o); } };
struct WavB { const float* ecT;             // [32 col][64 k]
  __device__ void load8(int, int k0, int col, float o[8]) const {
    ld8(ecT + ((size_t)col << 6) + k0, o); } };
struct WavC { float *d, *xn; int logNh;
  __device__ void store_tile(int, int rowb, int hi8, int col,
                             const v8f& a) const {
    int mh = (1 << logNh) - 1;
    float* p = (col < 16) ? d : xn;          // hoisted pointer select
    int cc = col & 15;
    int t0 = (rowb + hi8) >> 2;              // rowb+hi8 is a multiple of 8
    size_t pb[2];
#pragma unroll
    for (int h = 0; h < 2; ++h) {
      int t = t0 + h;
      int yo = t & mh; int t2 = t >> logNh; int xo = t2 & mh;
      int b = t2 >> logNh;
      pb[h] = ((((((size_t)b << logNh) + xo) << logNh) + yo) << 6) + cc;
    }
#pragma unroll
    for (int r = 0; r < 8; ++r)
      p[pb[r >> 2] + ((size_t)(r & 3) << 4)] = a[r]; } };

// Conv3x3 implicit GEMM: K = 9 taps x 64 ci = 576; one mask per 8-run
struct ConvA { const float* h; int logN;
  __device__ void load8(int, int row, int k0, float o[8]) const {
    int N = 1 << logN, mk = N - 1;
    int y = row & mk; int x = (row >> logN) & mk; int b = row >> (2*logN);
    int tap = k0 >> 6, ci0 = k0 & 63;
    int xx = x + tap/3 - 1, yy = y + tap%3 - 1;
    bool ok = ((unsigned)xx < (unsigned)N) && ((unsigned)yy < (unsigned)N);
    int xc = ok ? xx : 0, yc = ok ? yy : 0;
    size_t pix = ((((size_t)b << logN) + xc) << logN) + yc;
    ld8(h + (pix << 6) + ci0, o);
    float m = ok ? 1.f : 0.f;
#pragma unroll
    for (int i = 0; i < 8; ++i) o[i] *= m; } };
struct ConvB { const float* wT;             // [64 o][576 k]
  __device__ void load8(int, int k0, int col, float o[8]) const {
    ld8(wT + (size_t)col*576 + k0, o); } };
struct ConvC { float* out; const float* cb;
  __device__ void store_tile(int, int rowb, int hi8, int col,
                             const v8f& a) const {
    size_t base = ((size_t)(rowb + hi8) << 6) + col;
    float bv = cb[col];
#pragma unroll
    for (int r = 0; r < 8; ++r) out[base + ((size_t)r << 6)] = a[r] + bv; } };

// Linear 64->64 (w[out,in] is column-major per o); clamp floor = 0 (ReLU)
// or -FLT_MAX (identity) -> branchless v_max.
struct LinA { const float* in; float lob;
  __device__ void load8(int, int row, int k0, float o[8]) const {
    ld8(in + ((size_t)row << 6) + k0, o);
#pragma unroll
    for (int i = 0; i < 8; ++i) o[i] = fmaxf(o[i], lob); } };
struct LinB { const float* w;
  __device__ void load8(int, int k0, int col, float o[8]) const {
    ld8(w + ((size_t)col << 6) + k0, o); } };
struct LinC { float* out; const float* bias; int accum;
  __device__ void store_tile(int, int rowb, int hi8, int col,
                             const v8f& a) const {
    size_t base = ((size_t)(rowb + hi8) << 6) + col;
    float bv = bias[col];
    if (accum) {
#pragma unroll
      for (int r = 0; r < 8; ++r) {
        size_t i = base + ((size_t)r << 6); out[i] = out[i] + a[r] + bv; }
    } else {
#pragma unroll
      for (int r = 0; r < 8; ++r) out[base + ((size_t)r << 6)] = a[r] + bv;
    } } };

// Forward partial DFT: Xf[b,kk,ch] = sum_p afwd[kk,p] * dT[b,ch,p]
struct FwdA { const float* afwd; int NNp;   // [Mp rows][NNp], zero-padded
  __device__ void load8(int, int row, int k0, float o[8]) const {
    ld8(afwd + (size_t)row*NNp + k0, o); } };
struct FwdB { const float* dT; int NNp;     // [b][64 ch][NNp], zero-padded
  __device__ void load8(int b, int k0, int col, float o[8]) const {
    ld8(dT + ((size_t)((b << 6) + col))*NNp + k0, o); } };
struct FwdC { float* Xf; int Mp;            // [b][Mp][64]
  __device__ void store_tile(int b, int rowb, int hi8, int col,
                             const v8f& a) const {
    size_t base = (((size_t)b*Mp + rowb + hi8) << 6) + col;
#pragma unroll
    for (int r = 0; r < 8; ++r) Xf[base + ((size_t)r << 6)] = a[r]; } };

// Spectral mix per mode m: [Xre Xim](32x128) * Wp[m](128x128, col-major)
struct SpecA { const float* Xf; int Mp, Mm;
  __device__ void load8(int m, int b, int k0, float o[8]) const {
    int pr = k0 >> 6, i0 = k0 & 63;
    ld8(Xf + (((size_t)b*Mp + (size_t)pr*Mm + m) << 6) + i0, o); } };
struct SpecBW { const float* Wp;            // [m][128 nn][128 kk]
  __device__ void load8(int m, int k0, int col, float o[8]) const {
    ld8(Wp + (((size_t)m*128 + col) << 7) + k0, o); } };
struct SpecC { float* OutbT; int KKp, Mm;   // [b][64 o][KKp]
  __device__ void store_tile(int m, int rowb, int hi8, int col,
                             const v8f& a) const {
    int po = col >> 6, o = col & 63;
    size_t step = (size_t)KKp << 6;         // advance one b
    size_t base = ((size_t)(((rowb + hi8) << 6) + o))*KKp
                  + (size_t)po*Mm + m;
#pragma unroll
    for (int r = 0; r < 8; ++r) OutbT[base + (size_t)r*step] = a[r]; } };

// Inverse partial DFT: hft[b,p,o] = sum_kk ainv[p,kk] * OutbT[b,o,kk]
struct InvA { const float* ainv; int KKp;   // [NNr rows][KKp], zero-padded
  __device__ void load8(int, int row, int k0, float o[8]) const {
    ld8(ainv + (size_t)row*KKp + k0, o); } };
struct InvB { const float* OutbT; int KKp;
  __device__ void load8(int b, int k0, int col, float o[8]) const {
    ld8(OutbT + ((size_t)((b << 6) + col))*KKp + k0, o); } };
struct InvC { float* hft; int NN;
  __device__ void store_tile(int b, int rowb, int hi8, int col,
                             const v8f& a) const {
    size_t base = (((size_t)b*NN + rowb + hi8) << 6) + col;
    if (rowb + 16 <= NN) {                   // uniform full-tile fast path
#pragma unroll
      for (int r = 0; r < 8; ++r) hft[base + ((size_t)r << 6)] = a[r];
    } else {
#pragma unroll
      for (int r = 0; r < 8; ++r)
        if (rowb + hi8 + r < NN) hft[base + ((size_t)r << 6)] = a[r];
    } } };

// Synthesis: A = [x+Us | Ud] (K=32), B = rcT (col-major), scatter to 2Nx2N
struct RecA { const float *x, *Us, *Ud; int logN;
  __device__ void load8(int, int row, int k0, float o[8]) const {
    int mk = (1 << logN) - 1;
    int c = row & 3; int t = row >> 2;
    int yo = t & mk; t >>= logN; int xo = t & mk; int b = t >> logN;
    size_t pix = ((((size_t)b << logN) + xo) << logN) + yo;
    size_t base = (pix << 6) + (c << 4);
    if (k0 < 16) {                           // uniform at runtime
      float u[8]; ld8(x + base + k0, o); ld8(Us + base + k0, u);
#pragma unroll
      for (int i = 0; i < 8; ++i) o[i] += u[i];
    } else ld8(Ud + base + (k0 - 16), o); } };
struct RecB { const float* rcT;             // [64 col][32 k]
  __device__ void load8(int, int k0, int col, float o[8]) const {
    ld8(rcT + ((size_t)col << 5) + k0, o); } };
struct RecC { float* out; int logN;
  __device__ void store_tile(int, int rowb, int hi8, int col,
                             const v8f& a) const {
    int mk = (1 << logN) - 1;
    int q = col >> 4, j = col & 15;
    int t0 = (rowb + hi8) >> 2;
    size_t pb[2];
#pragma unroll
    for (int h = 0; h < 2; ++h) {
      int t = t0 + h;
      int yo = t & mk; int t2 = t >> logN; int xo = t2 & mk;
      int b = t2 >> logN;
      int xx = 2*xo + (q >> 1), yy = 2*yo + (q & 1);
      pb[h] = (((((size_t)b << (logN+1)) + xx) << (logN+1)) + yy) << 6;
    }
#pragma unroll
    for (int r = 0; r < 8; ++r)
      out[pb[r >> 2] + ((size_t)(r & 3) << 4) + j] = a[r]; } };

// ----------------------- staging / table kernels ---------------------------
__device__ inline float twval(int kk, int p, int N, int logN, int l1, int l2,
                              int nR, int Mm, int* ky_out) {
  int part = (kk >= Mm); int m = part ? kk - Mm : kk;
  int j = m / l2, ky = m - j*l2;
  int r = (j < l1) ? j : (N - nR + j);
  int x = p >> logN, y = p & (N - 1);
  int ph = (r*x + ky*y) & (N - 1);
  float th = 6.28318530717958647692f * (float)ph / (float)N;
  float s, c; __sincosf(th, &s, &c);
  *ky_out = ky;
  return part ? -s : c;
}

// afwd: [Mp][NNp] (cos | -sin, zero pad); ainv: [NNr][KKp] (w/N^2 folded, pad)
__global__ void gen_tw(float* afwd, float* ainv, int N, int logN, int NN,
                       int l1, int l2, int nR, int Mm,
                       int Mp, int NNp, int NNr, int KKp) {
  int twoMm = 2*Mm;
  long tot1 = (long)Mp*NNp, tot2 = (long)NNr*KKp;
  for (long idx = (long)blockIdx.x*blockDim.x + threadIdx.x;
       idx < tot1 + tot2; idx += (long)gridDim.x*blockDim.x) {
    if (idx < tot1) {
      int p = (int)(idx % NNp); int kk = (int)(idx / NNp);
      float v = 0.f; int ky;
      if (p < NN && kk < twoMm) v = twval(kk, p, N, logN, l1, l2, nR, Mm, &ky);
      afwd[idx] = v;
    } else {
      long j2 = idx - tot1;
      int kk = (int)(j2 % KKp); int p = (int)(j2 / KKp);
      float v = 0.f;
      if (p < NN && kk < twoMm) {
        int ky; float c = twval(kk, p, N, logN, l1, l2, nR, Mm, &ky);
        float w = (ky == 0 || 2*ky == N) ? 1.f : 2.f;
        v = c * w / (float)NN;
      }
      ainv[j2] = v;
    }
  }
}

// dT[b][ch][p] (zero-padded p) from d[b][p][ch]
__global__ void tr_d(const float* d, float* dT, int NN, int logNNp) {
  long total = (long)32*64 << logNNp;
  int mp = (1 << logNNp) - 1;
  for (long idx = (long)blockIdx.x*blockDim.x + threadIdx.x; idx < total;
       idx += (long)gridDim.x*blockDim.x) {
    int p = (int)(idx & mp); long bc = idx >> logNNp;
    int ch = (int)(bc & 63); int b = (int)(bc >> 6);
    dT[idx] = (p < NN) ? d[(((size_t)b*NN + p) << 6) + ch] : 0.f;
  }
}

// Wp[m][nn][kk]: real 128x128 block form [[Wre Wim],[-Wim Wre]] per mode
__global__ void pack_wp(const float* w1r, const float* w1i,
                        const float* w2r, const float* w2i, float* Wp,
                        int N, int l1, int l2, int nR, int Mm) {
  long total = (long)Mm*128*128;
  for (long idx = (long)blockIdx.x*blockDim.x + threadIdx.x; idx < total;
       idx += (long)gridDim.x*blockDim.x) {
    int kk = (int)(idx & 127); long t = idx >> 7;
    int nn = (int)(t & 127); int m = (int)(t >> 7);
    int j = m / l2, ky = m - j*l2;
    int r = (j < l1) ? j : (N - nR + j);
    bool u2 = (r >= N - l1);                 // torch overwrite rule
    int kxw = u2 ? (r - (N - l1)) : r;
    int i = kk & 63, pr = kk >> 6, o = nn & 63, po = nn >> 6;
    size_t base = ((((size_t)i << 6) + o)*144) + (size_t)kxw*12 + ky;
    float re = u2 ? w2r[base] : w1r[base];
    float im = u2 ? w2i[base] : w1i[base];
    Wp[idx] = (pr == 0) ? ((po == 0) ? re : im) : ((po == 0) ? -im : re);
  }
}

__global__ void pack_ec(const float* ecd, const float* ecs, float* ecT) {
  int t = threadIdx.x + blockIdx.x*blockDim.x;      // 32*64
  if (t >= 32*64) return;
  int col = t >> 6, k = t & 63;
  ecT[t] = (col < 16) ? ecd[k*16 + col] : ecs[k*16 + col - 16];
}
__global__ void pack_rc(const float* ee, const float* eo, const float* oe,
                        const float* oo, float* rcT) {
  int t = threadIdx.x + blockIdx.x*blockDim.x;      // 64*32
  if (t >= 64*32) return;
  int col = t >> 5, k = t & 31;
  int q = col >> 4, j = col & 15;
  const float* rc = (q == 0) ? ee : (q == 1) ? eo : (q == 2) ? oe : oo;
  rcT[t] = rc[k*16 + j];
}
__global__ void pack_conv(const float* w, float* wT) {
  int t = threadIdx.x + blockIdx.x*blockDim.x;      // 64*576
  if (t >= 64*576) return;
  int o = t / 576, k = t - o*576;
  int tap = k >> 6, ci = k & 63;
  wT[t] = w[(((size_t)o << 6) + ci)*9 + tap];
}
__global__ void fillz(float* p, long n) {
  for (long i = (long)blockIdx.x*blockDim.x + threadIdx.x; i < n;
       i += (long)gridDim.x*blockDim.x) p[i] = 0.f;
}

// ------------------------------- host --------------------------------------
template<class A, class B, class C>
static inline void run_gemm(hipStream_t s, int batches, int Mb, int K,
                            int ntiles, A a, B b, C c) {
  int mtiles = (Mb + 15) >> 4;
  dim3 grid((unsigned)((mtiles + 7) >> 3), (unsigned)ntiles, (unsigned)batches);
  wmma_gemm<<<grid, 256, 0, s>>>(mtiles, K, a, b, c);
}
static inline int imin(int a, int b) { return a < b ? a : b; }
static inline int ru16(int v) { return (v + 15) & ~15; }
static inline int ru32(int v) { return (v + 31) & ~31; }
static inline int ilog2(int v) { int l = 0; while ((1 << l) < v) ++l; return l; }

extern "C" void kernel_launch(void* const* d_in, const int*, int,
                              void* d_out, int, void* d_ws, size_t ws_size,
                              hipStream_t stream) {
  const float* X0  = (const float*)d_in[0];
  const float* W1R = (const float*)d_in[1];  const float* W1I = (const float*)d_in[2];
  const float* W2R = (const float*)d_in[3];  const float* W2I = (const float*)d_in[4];
  const float* ALW = (const float*)d_in[5];  const float* ALB = (const float*)d_in[6];
  const float* BCW = (const float*)d_in[7];  const float* BCB = (const float*)d_in[8];
  const float* BLW = (const float*)d_in[9];  const float* BLB = (const float*)d_in[10];
  const float* CCW = (const float*)d_in[11]; const float* CCB = (const float*)d_in[12];
  const float* CLW = (const float*)d_in[13]; const float* CLB = (const float*)d_in[14];
  const float* T0W = (const float*)d_in[15]; const float* T0B = (const float*)d_in[16];
  const float* ECS = (const float*)d_in[17]; const float* ECD = (const float*)d_in[18];
  const float* REE = (const float*)d_in[19]; const float* REO = (const float*)d_in[20];
  const float* ROE = (const float*)d_in[21]; const float* ROO = (const float*)d_in[22];

  const int Bb = 32, NL = 7;
  const float NORELU = -3.402823466e38f, RELU = 0.f;
  int Ndl[NL], NNl[NL], l1l[NL], nRl[NL], Mml[NL];
  int Mpl[NL], NNpl[NL], NNrl[NL], KKpl[NL];
  for (int l = 0; l < NL; ++l) {
    int Nd = 64 >> l; Ndl[l] = Nd; NNl[l] = Nd*Nd;
    int l1 = imin(12, Nd/2 + 1); l1l[l] = l1;
    nRl[l] = imin(Nd, 2*l1); Mml[l] = nRl[l]*l1;
    Mpl[l]  = ru16(2*Mml[l]);  NNpl[l] = ru32(NNl[l]);
    NNrl[l] = ru16(NNl[l]);    KKpl[l] = ru32(2*Mml[l]);
  }

  char* ws = (char*)d_ws; size_t off = 0;
  auto alloc = [&](size_t nflt) {
    float* p = (float*)(ws + off);
    off += nflt*4; off = (off + 255) & ~(size_t)255; return p; };

  float *Ud[NL], *Us[NL], *AF[NL], *AI[NL];
  for (int l = 0; l < NL; ++l) {
    Ud[l] = alloc((size_t)Bb*NNl[l]*64);
    Us[l] = alloc((size_t)Bb*NNl[l]*64);
    AF[l] = alloc((size_t)Mpl[l]*NNpl[l]);
    AI[l] = alloc((size_t)NNrl[l]*KKpl[l]);
  }
  size_t big = (size_t)Bb*64*64*64;
  float* dbuf  = alloc(big);
  float* xb0   = alloc(big);
  float* xb1   = alloc(big);
  float* tmp   = alloc(big);
  float* dT    = alloc((size_t)Bb*64*NNpl[0]);
  float* Xf    = alloc((size_t)Bb*Mpl[0]*64);
  float* OutbT = alloc((size_t)Bb*64*KKpl[0]);
  float* Wp    = alloc((size_t)Mml[0]*128*128);
  float* recA  = alloc(big);
  float* recB  = alloc(big);
  float* ecT   = alloc(32*64);
  float* rcT   = alloc(64*32);
  float* cTB   = alloc(64*576);
  float* cTC   = alloc(64*576);
  if (off > ws_size) return;   // workspace too small: bail deterministically

  // one-time (per call) weight staging + twiddle tables
  pack_ec<<<8, 256, 0, stream>>>(ECD, ECS, ecT);
  pack_rc<<<8, 256, 0, stream>>>(REE, REO, ROE, ROO, rcT);
  pack_conv<<<144, 256, 0, stream>>>(BCW, cTB);
  pack_conv<<<144, 256, 0, stream>>>(CCW, cTC);
  for (int l = 0; l < NL; ++l) {
    long total = (long)Mpl[l]*NNpl[l] + (long)NNrl[l]*KKpl[l];
    int blocks = (int)((total + 255) / 256);
    gen_tw<<<blocks, 256, 0, stream>>>(AF[l], AI[l], Ndl[l], ilog2(Ndl[l]),
                                       NNl[l], l1l[l], l1l[l], nRl[l], Mml[l],
                                       Mpl[l], NNpl[l], NNrl[l], KKpl[l]);
  }

  // analysis sweep
  const float* xcur = X0; int logNin = 7;
  for (int l = 0; l < NL; ++l) {
    int Nd = Ndl[l], NN = NNl[l], l1 = l1l[l], nR = nRl[l], Mm = Mml[l];
    int Mp = Mpl[l], NNp = NNpl[l], KKp = KKpl[l];
    int logNd = ilog2(Nd);
    float* xnext = (l & 1) ? xb1 : xb0;
    // wavelet: [d | x_next] = gather(x) @ [ec_d | ec_s]
    run_gemm(stream, 1, Bb*NN*4, 64, 2,
             WavA{xcur, logNin, logNd}, WavB{ecT}, WavC{dbuf, xnext, logNd});
    // spectral path on d
    {
      long n = (long)Bb*64*NNp;
      tr_d<<<(int)((n + 255)/256), 256, 0, stream>>>(dbuf, dT, NN, ilog2(NNp));
    }
    run_gemm(stream, Bb, Mp, NNp, 4,
             FwdA{AF[l], NNp}, FwdB{dT, NNp}, FwdC{Xf, Mp});
    {
      long n = (long)Mm*128*128;
      pack_wp<<<(int)((n + 255)/256), 256, 0, stream>>>(
          W1R, W1I, W2R, W2I, Wp, Nd, l1, l1, nR, Mm);
    }
    if (2*Mm < KKp) {   // zero the padded kk tail once, SpecC fills the rest
      long n = (long)Bb*64*KKp;
      fillz<<<(int)((n + 255)/256), 256, 0, stream>>>(OutbT, n);
    }
    run_gemm(stream, Mm, 32, 128, 8,
             SpecA{Xf, Mp, Mm}, SpecBW{Wp}, SpecC{OutbT, KKp, Mm});
    run_gemm(stream, Bb, NN, KKp, 4,
             InvA{AI[l], KKp}, InvB{OutbT, KKp}, InvC{tmp, NN});
    run_gemm(stream, 1, Bb*NN, 64, 4,
             LinA{tmp, RELU}, LinB{ALW}, LinC{Ud[l], ALB, 0});  // Ud = FT part
    // sparse kernel B on x_next, accumulated into Ud
    run_gemm(stream, 1, Bb*NN, 576, 4,
             ConvA{xnext, logNd}, ConvB{cTB}, ConvC{tmp, BCB});
    run_gemm(stream, 1, Bb*NN, 64, 4,
             LinA{tmp, RELU}, LinB{BLW}, LinC{Ud[l], BLB, 1});  // Ud += conv
    // sparse kernel C on d -> Us
    run_gemm(stream, 1, Bb*NN, 576, 4,
             ConvA{dbuf, logNd}, ConvB{cTC}, ConvC{tmp, CCB});
    run_gemm(stream, 1, Bb*NN, 64, 4,
             LinA{tmp, RELU}, LinB{CLW}, LinC{Us[l], CLB, 0});
    xcur = xnext; logNin = logNd;
  }

  // coarsest-level linear T0 (no relu)
  run_gemm(stream, 1, Bb, 64, 4,
           LinA{xcur, NORELU}, LinB{T0W}, LinC{recA, T0B, 0});

  // synthesis sweep
  float* rin = recA;
  for (int i = NL - 1; i >= 0; --i) {
    int logN = ilog2(Ndl[i]);
    float* rout = (i == 0) ? (float*)d_out : ((rin == recA) ? recB : recA);
    run_gemm(stream, 1, Bb*NNl[i]*4, 32, 4,
             RecA{rin, Us[i], Ud[i], logN}, RecB{rcT}, RecC{rout, logN});
    rin = rout;
  }
}
